// Model_58325655880190
// MI455X (gfx1250) — matrix operand order repaired and checked by reading, not compiled, not run on hardware
//
#include <hip/hip_runtime.h>
#include <hip/hip_bf16.h>

typedef _Float16 v16h __attribute__((ext_vector_type(16)));
typedef _Float16 v8h  __attribute__((ext_vector_type(8)));
typedef float    v8f  __attribute__((ext_vector_type(8)));

#define NB    128      // batch
#define SEQ   1024     // sequence length
#define EMBD  300      // embedding dim
#define EMBP  320      // padded embedding dim (10 k-tiles of 32)
#define CPAD  128      // padded channel dim (4 k-tiles of 32)

static __device__ __forceinline__ v16h mk16(v8h lo, v8h hi) {
  return __builtin_shufflevector(lo, hi, 0,1,2,3,4,5,6,7,8,9,10,11,12,13,14,15);
}

// ---------------------------------------------------------------------------
// Kernel 0: repack conv weights to f16 [tap][Npad=128][Kpad] (B-operand layout)
// and zero the global-max accumulator.
// ---------------------------------------------------------------------------
__global__ void prep_kernel(const float* __restrict__ w1, const float* __restrict__ wl1,
                            const float* __restrict__ wl2, const float* __restrict__ wf1,
                            const float* __restrict__ wf2,
                            _Float16* __restrict__ w1t, _Float16* __restrict__ wl1t,
                            _Float16* __restrict__ wl2t, _Float16* __restrict__ wf1loc,
                            float* __restrict__ wf2pad, float* __restrict__ gpre) {
  int tid = blockIdx.x * blockDim.x + threadIdx.x;
  int nth = gridDim.x * blockDim.x;
  // w1t: [3][128][320] <- w1 (100,300,3) (o*900 + c*3 + k)
  for (int i = tid; i < 3 * 128 * EMBP; i += nth) {
    int k = i / (128 * EMBP); int o = (i / EMBP) & 127; int c = i % EMBP;
    float v = (o < 100 && c < EMBD) ? w1[o * 900 + c * 3 + k] : 0.f;
    w1t[i] = (_Float16)v;
  }
  // wl1t/wl2t: [3][128][128] <- (100,100,3) (o*300 + c*3 + k)
  for (int i = tid; i < 3 * 128 * CPAD; i += nth) {
    int k = i / (128 * CPAD); int o = (i / CPAD) & 127; int c = i % CPAD;
    bool in = (o < 100 && c < 100);
    wl1t[i] = (_Float16)(in ? wl1[o * 300 + c * 3 + k] : 0.f);
    wl2t[i] = (_Float16)(in ? wl2[o * 300 + c * 3 + k] : 0.f);
  }
  // wf1loc: [128][128] <- wf1 (100,200,1), local half: c in [100,200)
  for (int i = tid; i < 128 * CPAD; i += nth) {
    int o = i / CPAD; int c = i % CPAD;
    wf1loc[i] = (_Float16)((o < 100 && c < 100) ? wf1[o * 200 + 100 + c] : 0.f);
  }
  for (int i = tid; i < 128; i += nth) wf2pad[i] = (i < 100) ? wf2[i] : 0.f;
  for (int i = tid; i < NB * 128; i += nth) gpre[i] = 0.f;
}

// ---------------------------------------------------------------------------
// Kernel 1: gather emb[x] -> f16, channel-padded to 320 (one wave per token)
// ---------------------------------------------------------------------------
__global__ void gather_kernel(const int* __restrict__ x, const float* __restrict__ emb,
                              _Float16* __restrict__ xe) {
  int wid  = blockIdx.x * (blockDim.x >> 5) + (threadIdx.x >> 5);
  int lane = threadIdx.x & 31;
  if (wid >= NB * SEQ) return;
  int row = x[wid];
  const float* src = emb + (size_t)row * EMBD;
  _Float16* dst = xe + (size_t)wid * EMBP;
  for (int c = lane; c < EMBP; c += 32)
    dst[c] = (c < EMBD) ? (_Float16)src[c] : (_Float16)0.f;
}

// ---------------------------------------------------------------------------
// WMMA conv layer: out[b][h][128] = relu(bias + sum_{tap,c} in[b][h+tap-1][c] * W)
// One wave computes a 16(h) x 128(ch) tile via v_wmma_f32_16x16x32_f16.
// ---------------------------------------------------------------------------
template <int KT, int TAPS, bool DOMAX>
__global__ void conv_wmma_kernel(const _Float16* __restrict__ in, int istride,
                                 const _Float16* __restrict__ wt,
                                 const float* __restrict__ bias,
                                 _Float16* __restrict__ out,
                                 float* __restrict__ gmax) {
  const int wid = blockIdx.x * (blockDim.x >> 5) + (threadIdx.x >> 5);
  if (wid >= NB * (SEQ / 16)) return;
  const int b    = wid >> 6;
  const int h0   = (wid & 63) << 4;
  const int lane = threadIdx.x & 31;
  const int lr   = lane & 15;
  const int half = lane >> 4;
  const int Kdim = KT * 32;
  const int kaoff = half * 8;    // A: K halves {kaoff..+7} U {kaoff+16..+23}
  const int kboff = half * 8;    // B: K halves {kboff..+7} U {kboff+16..+23}
  v8f acc[8] = {};
  v8h z8 = {};
  for (int kt = 0; kt < KT; ++kt) {
    const int kbase = kt * 32;
    v16h afr[TAPS];
#pragma unroll
    for (int t = 0; t < TAPS; ++t) {
      int row = h0 + lr + t - (TAPS > 1 ? 1 : 0);
      bool valid = (row >= 0) && (row < SEQ);
      int rc = row < 0 ? 0 : (row > SEQ - 1 ? SEQ - 1 : row);
      const _Float16* ap = in + ((size_t)b * SEQ + rc) * istride + kbase + kaoff;
      v8h lo = *(const v8h*)ap;
      v8h hi = *(const v8h*)(ap + 16);
      if (!valid) { lo = z8; hi = z8; }
      afr[t] = mk16(lo, hi);
    }
#pragma unroll
    for (int nt = 0; nt < 8; ++nt) {
#pragma unroll
      for (int t = 0; t < TAPS; ++t) {
        const _Float16* bp = wt + ((size_t)t * 128 + nt * 16 + lr) * Kdim + kbase + kboff;
        v8h blo = *(const v8h*)bp;
        v8h bhi = *(const v8h*)(bp + 16);
        v16h bfr = mk16(blo, bhi);
        acc[nt] = __builtin_amdgcn_wmma_f32_16x16x32_f16(
            false, afr[t], false, bfr, (short)0, acc[nt], false, false);
      }
    }
  }
#pragma unroll
  for (int nt = 0; nt < 8; ++nt) {
    int n = nt * 16 + lr;
    float bv = (n < 100) ? bias[n] : 0.f;
    float mx = 0.f;
#pragma unroll
    for (int r = 0; r < 8; ++r) {
      float v = fmaxf(acc[nt][r] + bv, 0.f);
      out[((size_t)b * SEQ + h0 + half * 8 + r) * CPAD + n] = (_Float16)v;
      mx = fmaxf(mx, v);
    }
    if (DOMAX)  // post-ReLU values are >= 0: integer max == float max
      atomicMax((int*)(gmax + b * 128 + n), __float_as_int(mx));
  }
}

// ---------------------------------------------------------------------------
// Kernel: g = relu(maxpool @ wg.T + bg); fbase[b][o] = bf1[o] + g . wf1[:,0:100]
// ---------------------------------------------------------------------------
__global__ void g_fbase_kernel(const float* __restrict__ gpre, const float* __restrict__ wg,
                               const float* __restrict__ bg, const float* __restrict__ wf1,
                               const float* __restrict__ bf1, float* __restrict__ fbase) {
  int b = blockIdx.x; int o = threadIdx.x;  // 128 threads
  __shared__ float gs[100];
  if (o < 100) {
    float a = bg[o];
    const float* gp = gpre + b * 128;
    const float* wr = wg + o * 100;
    for (int c = 0; c < 100; ++c) a += gp[c] * wr[c];
    gs[o] = fmaxf(a, 0.f);
  }
  __syncthreads();
  float f = 0.f;
  if (o < 100) {
    f = bf1[o];
    const float* wr = wf1 + o * 200;
    for (int c = 0; c < 100; ++c) f += gs[c] * wr[c];
  }
  if (o < 128) fbase[b * 128 + o] = f;
}

// ---------------------------------------------------------------------------
// Kernel: fin = relu(fbase + loc2 @ wf1_loc.T); logits = fin . wf2 + bf2
// WMMA for the 128-K GEMM, cross-lane shfl reduction for the wf2 dot.
// ---------------------------------------------------------------------------
__global__ void fin_logits_kernel(const _Float16* __restrict__ loc2,
                                  const _Float16* __restrict__ wf1loc,
                                  const float* __restrict__ fbase,
                                  const float* __restrict__ wf2pad,
                                  const float* __restrict__ bf2,
                                  float* __restrict__ logits) {
  const int wid = blockIdx.x * (blockDim.x >> 5) + (threadIdx.x >> 5);
  if (wid >= NB * (SEQ / 16)) return;
  const int b    = wid >> 6;
  const int h0   = (wid & 63) << 4;
  const int lane = threadIdx.x & 31;
  const int lr   = lane & 15;
  const int half = lane >> 4;
  v8f acc[8] = {};
  for (int kt = 0; kt < 4; ++kt) {
    const int kbase = kt * 32;
    const _Float16* ap = loc2 + ((size_t)b * SEQ + h0 + lr) * CPAD + kbase + half * 8;
    v16h a = mk16(*(const v8h*)ap, *(const v8h*)(ap + 16));
#pragma unroll
    for (int nt = 0; nt < 8; ++nt) {
      const _Float16* bp = wf1loc + ((size_t)(nt * 16 + lr)) * CPAD + kbase + half * 8;
      v16h bm = mk16(*(const v8h*)bp, *(const v8h*)(bp + 16));
      acc[nt] = __builtin_amdgcn_wmma_f32_16x16x32_f16(
          false, a, false, bm, (short)0, acc[nt], false, false);
    }
  }
  float sr[8] = {0, 0, 0, 0, 0, 0, 0, 0};
#pragma unroll
  for (int nt = 0; nt < 8; ++nt) {
    int n = nt * 16 + lr;
    float fb = fbase[b * 128 + n];
    float w2 = wf2pad[n];
#pragma unroll
    for (int r = 0; r < 8; ++r)
      sr[r] += fmaxf(fb + acc[nt][r], 0.f) * w2;
  }
#pragma unroll
  for (int r = 0; r < 8; ++r) {
    float s = sr[r];
    s += __shfl_xor(s, 1, 32);
    s += __shfl_xor(s, 2, 32);
    s += __shfl_xor(s, 4, 32);
    s += __shfl_xor(s, 8, 32);   // reduce within each 16-lane half
    if (lr == 0)
      logits[(size_t)b * SEQ + h0 + half * 8 + r] = s + bf2[0];
  }
}

// ---------------------------------------------------------------------------
// Deterministic hash-based approximation of the sum-of-gammas Gumbel noise.
// ---------------------------------------------------------------------------
static __device__ __forceinline__ unsigned hash_u32(unsigned s) {
  s ^= s >> 16; s *= 0x7feb352dU; s ^= s >> 15; s *= 0x846ca68bU; s ^= s >> 16;
  return s;
}
static __device__ __forceinline__ float u01(unsigned h) {
  return (h >> 8) * (1.0f / 16777216.0f) + 5.96e-8f;
}
static __device__ __forceinline__ float gumbel_noise(int b, int h) {
  unsigned base = (unsigned)(b * SEQ + h) * 0x9E3779B9u + 0x42424242u;
  float sog = 0.f;
#pragma unroll
  for (int t = 1; t <= 10; ++t) {
    float uA = u01(hash_u32(base + (unsigned)t * 0x68bc21ebU));
    float uB = u01(hash_u32(base + (unsigned)t * 0x02e1b213U + 0x1234567u));
    // Gamma(1/K): Gamma(a) = Gamma(1+a) * U^(1/a), Gamma(1.1) ~ Exp(1)
    float g = -logf(uA) * powf(uB, 10.0f);
    sog += g * (10.0f / (float)t);
  }
  sog -= 2.302585093f;       // log(N_GAMMA)
  return sog;                // TAU * sog / K with TAU=K=10
}

// ---------------------------------------------------------------------------
// Kernel: per-batch top-10 threshold, masked embedding sum, final MLP+sigmoid.
// One 1024-thread block per batch row.
// ---------------------------------------------------------------------------
__global__ void final_kernel(const float* __restrict__ logits, const int* __restrict__ x,
                             const float* __restrict__ emb, const float* __restrict__ wp1,
                             const float* __restrict__ bp1, const float* __restrict__ wp2,
                             const float* __restrict__ bp2, float* __restrict__ out) {
  int b = blockIdx.x, tid = threadIdx.x;
  __shared__ float gs[SEQ];
  __shared__ float red_v[32];
  __shared__ int   red_i[32];
  __shared__ float thr_s;
  __shared__ int   sel[64];
  __shared__ int   cnt;
  __shared__ float res[EMBD];
  __shared__ float r2[200];
  float g0 = logits[b * SEQ + tid] + gumbel_noise(b, tid);
  gs[tid] = g0;
  if (tid == 0) cnt = 0;
  __syncthreads();
  for (int it = 0; it < 10; ++it) {
    float v = gs[tid]; int idx = tid;
#pragma unroll
    for (int off = 16; off >= 1; off >>= 1) {
      float ov = __shfl_xor(v, off, 32);
      int   oi = __shfl_xor(idx, off, 32);
      if (ov > v) { v = ov; idx = oi; }
    }
    if ((tid & 31) == 0) { red_v[tid >> 5] = v; red_i[tid >> 5] = idx; }
    __syncthreads();
    if (tid < 32) {
      v = red_v[tid]; idx = red_i[tid];
#pragma unroll
      for (int off = 16; off >= 1; off >>= 1) {
        float ov = __shfl_xor(v, off, 32);
        int   oi = __shfl_xor(idx, off, 32);
        if (ov > v) { v = ov; idx = oi; }
      }
      if (tid == 0) { thr_s = v; gs[idx] = -3.0e38f; }
    }
    __syncthreads();
  }
  float thr = thr_s;
  if (g0 >= thr) {
    int p = atomicAdd(&cnt, 1);
    if (p < 64) sel[p] = tid;
  }
  __syncthreads();
  int n = cnt < 64 ? cnt : 64;
  if (tid < EMBD) {
    float a = 0.f;
    for (int i = 0; i < n; ++i) {
      int tok = x[b * SEQ + sel[i]];
      a += emb[(size_t)tok * EMBD + tid];
    }
    res[tid] = a * 0.1f;  // / K
  }
  __syncthreads();
  if (tid < 200) {
    float a = bp1[tid];
    const float* wr = wp1 + tid * EMBD;
    for (int c = 0; c < EMBD; ++c) a += res[c] * wr[c];
    r2[tid] = fmaxf(a, 0.f);
  }
  __syncthreads();
  if (tid == 0) {
    float a = bp2[0];
    for (int c = 0; c < 200; ++c) a += r2[c] * wp2[c];
    out[b] = 1.f / (1.f + expf(-a));
  }
}

// ---------------------------------------------------------------------------
extern "C" void kernel_launch(void* const* d_in, const int* in_sizes, int n_in,
                              void* d_out, int out_size, void* d_ws, size_t ws_size,
                              hipStream_t stream) {
  (void)in_sizes; (void)n_in; (void)out_size; (void)ws_size;
  const int*   x   = (const int*)  d_in[0];
  const float* emb = (const float*)d_in[1];
  const float* w1  = (const float*)d_in[2];
  const float* b1  = (const float*)d_in[3];
  const float* wg  = (const float*)d_in[4];
  const float* bg  = (const float*)d_in[5];
  const float* wl1 = (const float*)d_in[6];
  const float* bl1 = (const float*)d_in[7];
  const float* wl2 = (const float*)d_in[8];
  const float* bl2 = (const float*)d_in[9];
  const float* wf1 = (const float*)d_in[10];
  const float* bf1 = (const float*)d_in[11];
  const float* wf2 = (const float*)d_in[12];
  const float* bf2 = (const float*)d_in[13];
  const float* wp1 = (const float*)d_in[14];
  const float* bp1 = (const float*)d_in[15];
  const float* wp2 = (const float*)d_in[16];
  const float* bp2 = (const float*)d_in[17];
  float* out = (float*)d_out;

  char* ws = (char*)d_ws;
  constexpr size_t OFF_XE     = 0;                                  // f16 [128][1024][320]
  constexpr size_t OFF_FIRST  = OFF_XE     + (size_t)NB*SEQ*EMBP*2; // f16 [128][1024][128]
  constexpr size_t OFF_LOC1   = OFF_FIRST  + (size_t)NB*SEQ*CPAD*2;
  constexpr size_t OFF_LOC2   = OFF_LOC1   + (size_t)NB*SEQ*CPAD*2;
  constexpr size_t OFF_W1T    = OFF_LOC2   + (size_t)NB*SEQ*CPAD*2; // f16 [3][128][320]
  constexpr size_t OFF_WL1T   = OFF_W1T    + (size_t)3*128*EMBP*2;  // f16 [3][128][128]
  constexpr size_t OFF_WL2T   = OFF_WL1T   + (size_t)3*128*CPAD*2;
  constexpr size_t OFF_WF1L   = OFF_WL2T   + (size_t)3*128*CPAD*2;  // f16 [128][128]
  constexpr size_t OFF_WF2P   = OFF_WF1L   + (size_t)128*CPAD*2;    // f32 [128]
  constexpr size_t OFF_GPRE   = OFF_WF2P   + (size_t)128*4;         // f32 [128][128]
  constexpr size_t OFF_FBASE  = OFF_GPRE   + (size_t)NB*128*4;      // f32 [128][128]
  constexpr size_t OFF_LOGITS = OFF_FBASE  + (size_t)NB*128*4;      // f32 [128][1024]

  _Float16* xe     = (_Float16*)(ws + OFF_XE);
  _Float16* first  = (_Float16*)(ws + OFF_FIRST);
  _Float16* loc1   = (_Float16*)(ws + OFF_LOC1);
  _Float16* loc2   = (_Float16*)(ws + OFF_LOC2);
  _Float16* w1t    = (_Float16*)(ws + OFF_W1T);
  _Float16* wl1t   = (_Float16*)(ws + OFF_WL1T);
  _Float16* wl2t   = (_Float16*)(ws + OFF_WL2T);
  _Float16* wf1loc = (_Float16*)(ws + OFF_WF1L);
  float*    wf2pad = (float*)   (ws + OFF_WF2P);
  float*    gpre   = (float*)   (ws + OFF_GPRE);
  float*    fbase  = (float*)   (ws + OFF_FBASE);
  float*    lgts   = (float*)   (ws + OFF_LOGITS);

  // 0: repack weights (f16, padded, B-operand layout) + zero max accumulator
  prep_kernel<<<256, 256, 0, stream>>>(w1, wl1, wl2, wf1, wf2,
                                       w1t, wl1t, wl2t, wf1loc, wf2pad, gpre);
  // 1: gather + f16 convert
  gather_kernel<<<(NB * SEQ) / 8, 256, 0, stream>>>(x, emb, xe);
  // 2: conv1 (K=320, 3 taps) + channel max for global pool
  conv_wmma_kernel<10, 3, true><<<1024, 256, 0, stream>>>(xe, EMBP, w1t, b1, first, gpre);
  // 3: g and fbase
  g_fbase_kernel<<<NB, 128, 0, stream>>>(gpre, wg, bg, wf1, bf1, fbase);
  // 4,5: local convs (K=128, 3 taps)
  conv_wmma_kernel<4, 3, false><<<1024, 256, 0, stream>>>(first, CPAD, wl1t, bl1, loc1, nullptr);
  conv_wmma_kernel<4, 3, false><<<1024, 256, 0, stream>>>(loc1, CPAD, wl2t, bl2, loc2, nullptr);
  // 6: wf1 GEMM fused with wf2 dot -> logits
  fin_logits_kernel<<<1024, 256, 0, stream>>>(loc2, wf1loc, fbase, wf2pad, bf2, lgts);
  // 7: top-10 mask, selected-token embedding sum, final MLP + sigmoid
  final_kernel<<<NB, 1024, 0, stream>>>(lgts, x, emb, wp1, bp1, wp2, bp2, out);
}
